// LaminiIndex_66305705116279
// MI455X (gfx1250) — compile-verified, matrix operand order checked by reading, not run on hardware
//
#include <hip/hip_runtime.h>
#include <hip/hip_bf16.h>

// ---------------------------------------------------------------------------
// Flash-style attention over a 65536-entry embedding table (MI455X / gfx1250)
//   Q (4096x128) fp32, K/V (65536x128) fp32
//   out0 = softmax(Q K^T) K ; out1 = softmax(Q K^T) V
// Single pass, online softmax, bf16 WMMA (v_wmma_f32_16x16x32_bf16), f32 acc.
// Software-pipelined global->LDS staging, batched DS loads ahead of WMMA,
// VALU-pipe (permlane16) softmax reductions.
// ---------------------------------------------------------------------------

typedef __attribute__((ext_vector_type(16))) __bf16 v16bf;
typedef __attribute__((ext_vector_type(8)))  float  v8f;
typedef __attribute__((ext_vector_type(4)))  float  f32x4;

#define EMBED_D      128
#define NKEYS        65536
#define QROWS_TOTAL  4096          // BATCH * SEQ_LEN
#define WAVES_PER_WG 4
#define WG_THREADS   (WAVES_PER_WG * 32)
#define QTILE        16            // query rows per wave
#define CHUNK        32            // keys processed per main-loop iteration
#define KN_STRIDE    136           // u16 stride of row-major K chunk (128+8 pad)
#define KT_STRIDE    40            // u16 stride of transposed K/V chunk (32+8 pad)
#define P_STRIDE     40            // u16 stride of per-wave P staging buffer
#define SLOTS_PER_T  8             // CHUNK*(D/4) / WG_THREADS float4 slots/thread

union BfPack {                      // 32-byte pun: bf16x16 <-> 2x b128
    v16bf v;
    unsigned short u[16];
    unsigned int   w[8];
    f32x4 f4[2];
};

// packed fp32x2 -> bf16x2 (v_cvt_pk_bf16_f32 on gfx1250); lo in bits[15:0]
__device__ __forceinline__ unsigned int f2bf2(float lo, float hi) {
    __hip_bfloat162 h = __float22bfloat162_rn(make_float2(lo, hi));
    unsigned int r;
    __builtin_memcpy(&r, &h, sizeof(r));   // __hip_bfloat162 isn't trivially
    return r;                              // copyable -> bit_cast not allowed
}

// raw v_max_num_f32: skips the canonicalize self-max fmaxf() emits
__device__ __forceinline__ float maxf_raw(float a, float b) {
    float d;
    asm("v_max_num_f32 %0, %1, %2" : "=v"(d) : "v"(a), "v"(b));
    return d;
}

// xor-butterfly lane move within 16-lane rows, on the VALU pipe
// (v_permlane16_b32) instead of the DS pipe (ds_bpermute from __shfl_xor).
__device__ __forceinline__ float lane_xor(float x, unsigned s0, unsigned s1,
                                          int mask) {
#if __has_builtin(__builtin_amdgcn_permlane16)
    (void)mask;
    unsigned u = __builtin_bit_cast(unsigned, x);
    unsigned r = __builtin_amdgcn_permlane16(u, u, s0, s1, false, false);
    return __builtin_bit_cast(float, r);
#else
    (void)s0; (void)s1;
    return __shfl_xor(x, mask, 32);
#endif
}
// nibble-select tables for lane^k within each row of 16 (lanes0-7 / lanes8-15)
#define XSEL1_LO 0x67452301u
#define XSEL1_HI 0xEFCDAB89u
#define XSEL2_LO 0x54761032u
#define XSEL2_HI 0xDCFE98BAu
#define XSEL4_LO 0x32107654u
#define XSEL4_HI 0xBA98FEDCu
#define XSEL8_LO 0xFEDCBA98u
#define XSEL8_HI 0x76543210u

__device__ __forceinline__ float rowmax16(float t) {
    t = maxf_raw(t, lane_xor(t, XSEL1_LO, XSEL1_HI, 1));
    t = maxf_raw(t, lane_xor(t, XSEL2_LO, XSEL2_HI, 2));
    t = maxf_raw(t, lane_xor(t, XSEL4_LO, XSEL4_HI, 4));
    t = maxf_raw(t, lane_xor(t, XSEL8_LO, XSEL8_HI, 8));
    return t;
}
__device__ __forceinline__ float rowsum16(float t) {
    t += lane_xor(t, XSEL1_LO, XSEL1_HI, 1);
    t += lane_xor(t, XSEL2_LO, XSEL2_HI, 2);
    t += lane_xor(t, XSEL4_LO, XSEL4_HI, 4);
    t += lane_xor(t, XSEL8_LO, XSEL8_HI, 8);
    return t;
}

__global__ __launch_bounds__(WG_THREADS) void
lamini_attn_kernel(const float* __restrict__ Q,
                   const float* __restrict__ K,
                   const float* __restrict__ V,
                   float* __restrict__ outK,
                   float* __restrict__ outV)
{
    // K chunk, row-major bf16 (B operand of Q*K^T): conflict-free b128 reads
    __shared__ __align__(16) unsigned short sKn[CHUNK * KN_STRIDE];
    // K^T / V^T bf16 (B operands of P*K, P*V)
    __shared__ __align__(16) unsigned short sKT[EMBED_D * KT_STRIDE];
    __shared__ __align__(16) unsigned short sVT[EMBED_D * KT_STRIDE];
    // per-wave P staging (C-layout -> A-layout transpose)
    __shared__ __align__(16) unsigned short sP[WAVES_PER_WG * QTILE * P_STRIDE];

    const int tid  = threadIdx.x;
    const int wave = tid >> 5;
    const int lane = tid & 31;
    const int half = lane >> 4;      // 0: lanes 0-15, 1: lanes 16-31
    const int ln   = lane & 15;

    const int qtile = blockIdx.x * WAVES_PER_WG + wave;   // 0..255
    const int qbase = qtile * QTILE;

    // ---- Load Q tile (16x128) into WMMA A-layout bf16, pre-scaled by log2(e)
    // A layout (16x32 bf16): lane m = ln; element e<8 -> k = half*8+e,
    //                                    element e>=8 -> k = 16+half*8+(e-8)
    v16bf qa[4];
    {
        const float  lg2e = 1.44269504088896340736f;
        const float* qrow = Q + (size_t)(qbase + ln) * EMBED_D;
        #pragma unroll
        for (int s = 0; s < 4; ++s) {
            BfPack t;
            const int dlo = 32 * s + half * 8;
            const int dhi = dlo + 16;
            #pragma unroll
            for (int e2 = 0; e2 < 4; ++e2) {
                t.w[e2]     = f2bf2(qrow[dlo + 2*e2] * lg2e, qrow[dlo + 2*e2 + 1] * lg2e);
                t.w[4 + e2] = f2bf2(qrow[dhi + 2*e2] * lg2e, qrow[dhi + 2*e2 + 1] * lg2e);
            }
            qa[s] = t.v;
        }
    }

    // ---- f32 accumulators: 16 queries x 128 dims for K-out and V-out
    v8f accK[8], accV[8];
    #pragma unroll
    for (int j = 0; j < 8; ++j)
        #pragma unroll
        for (int c = 0; c < 8; ++c) { accK[j][c] = 0.0f; accV[j][c] = 0.0f; }

    // per-lane row stats: component c <-> query row (c + 8*half)
    float m_r[8], l_r[8];
    #pragma unroll
    for (int c = 0; c < 8; ++c) { m_r[c] = -1.0e30f; l_r[c] = 0.0f; }

    // ---- software pipeline: chunk n+1 lives in registers during chunk n
    f32x4 kreg[SLOTS_PER_T], vreg[SLOTS_PER_T];
    #pragma unroll
    for (int i = 0; i < SLOTS_PER_T; ++i) {
        const int slot = tid + WG_THREADS * i;
        const int key  = slot >> 5;
        const int d0   = (slot & 31) << 2;
        kreg[i] = *reinterpret_cast<const f32x4*>(K + (size_t)key * EMBED_D + d0);
        vreg[i] = *reinterpret_cast<const f32x4*>(V + (size_t)key * EMBED_D + d0);
    }

    // =====================================================================
    for (int n0 = 0; n0 < NKEYS; n0 += CHUNK) {
        __syncthreads();   // previous chunk fully consumed by all waves

        // ---- drain pipeline registers: bf16 convert + LDS (K row, K^T, V^T)
        #pragma unroll
        for (int i = 0; i < SLOTS_PER_T; ++i) {
            const int slot = tid + WG_THREADS * i;
            const int key  = slot >> 5;          // 0..31
            const int d0   = (slot & 31) << 2;   // 0,4,...,124
            const unsigned int kp0 = f2bf2(kreg[i][0], kreg[i][1]);
            const unsigned int kp1 = f2bf2(kreg[i][2], kreg[i][3]);
            const unsigned int vp0 = f2bf2(vreg[i][0], vreg[i][1]);
            const unsigned int vp1 = f2bf2(vreg[i][2], vreg[i][3]);
            *reinterpret_cast<uint2*>(&sKn[key * KN_STRIDE + d0]) = make_uint2(kp0, kp1);
            sKT[(d0 + 0) * KT_STRIDE + key] = (unsigned short)kp0;
            sKT[(d0 + 1) * KT_STRIDE + key] = (unsigned short)(kp0 >> 16);
            sKT[(d0 + 2) * KT_STRIDE + key] = (unsigned short)kp1;
            sKT[(d0 + 3) * KT_STRIDE + key] = (unsigned short)(kp1 >> 16);
            sVT[(d0 + 0) * KT_STRIDE + key] = (unsigned short)vp0;
            sVT[(d0 + 1) * KT_STRIDE + key] = (unsigned short)(vp0 >> 16);
            sVT[(d0 + 2) * KT_STRIDE + key] = (unsigned short)vp1;
            sVT[(d0 + 3) * KT_STRIDE + key] = (unsigned short)(vp1 >> 16);
        }
        __syncthreads();

        // ---- issue next chunk's global loads now; consumed next iteration,
        //      so their latency hides under this chunk's WMMA work.
        if (n0 + CHUNK < NKEYS) {
            const int nn = n0 + CHUNK;
            #pragma unroll
            for (int i = 0; i < SLOTS_PER_T; ++i) {
                const int slot = tid + WG_THREADS * i;
                const int key  = slot >> 5;
                const int d0   = (slot & 31) << 2;
                kreg[i] = *reinterpret_cast<const f32x4*>(
                    K + (size_t)(nn + key) * EMBED_D + d0);
                vreg[i] = *reinterpret_cast<const f32x4*>(
                    V + (size_t)(nn + key) * EMBED_D + d0);
            }
            if (n0 + 2 * CHUNK < NKEYS) {  // warm L2/WGP$ two chunks ahead
                const char* nk = (const char*)(K + (size_t)(n0 + 2 * CHUNK) * EMBED_D);
                const char* nv = (const char*)(V + (size_t)(n0 + 2 * CHUNK) * EMBED_D);
                __builtin_prefetch(nk + tid * 128, 0, 1);
                __builtin_prefetch(nv + tid * 128, 0, 1);
            }
        }

        // ---- S = Q * K_chunk^T : batch all 16 ds_load_b128, then 2x4 WMMA
        BfPack b[8];   // [0..3]: keys n0..+15 slabs, [4..7]: keys +16..+31
        {
            const unsigned short* krow0 = &sKn[ln * KN_STRIDE];
            const unsigned short* krow1 = &sKn[(16 + ln) * KN_STRIDE];
            #pragma unroll
            for (int s = 0; s < 4; ++s) {
                const int dlo = 32 * s + half * 8;
                b[s].f4[0]     = *reinterpret_cast<const f32x4*>(krow0 + dlo);
                b[s].f4[1]     = *reinterpret_cast<const f32x4*>(krow0 + dlo + 16);
                b[4 + s].f4[0] = *reinterpret_cast<const f32x4*>(krow1 + dlo);
                b[4 + s].f4[1] = *reinterpret_cast<const f32x4*>(krow1 + dlo + 16);
            }
        }
        v8f c0, c1;
        #pragma unroll
        for (int c = 0; c < 8; ++c) { c0[c] = 0.0f; c1[c] = 0.0f; }
        #pragma unroll
        for (int s = 0; s < 4; ++s) {   // two independent 4-deep chains
            c0 = __builtin_amdgcn_wmma_f32_16x16x32_bf16(false, qa[s], false, b[s].v,
                                                         (short)0, c0, false, false);
            c1 = __builtin_amdgcn_wmma_f32_16x16x32_bf16(false, qa[s], false, b[4 + s].v,
                                                         (short)0, c1, false, false);
        }

        // ---- online softmax (exp2 domain; Q was pre-scaled by log2 e)
        float p0[8], p1[8], scale[8];
        #pragma unroll
        for (int c = 0; c < 8; ++c) {
            const float t  = rowmax16(maxf_raw(c0[c], c1[c]));
            const float mn = maxf_raw(m_r[c], t);
            const float sc = __builtin_amdgcn_exp2f(m_r[c] - mn);
            const float a  = __builtin_amdgcn_exp2f(c0[c] - mn);
            const float bb = __builtin_amdgcn_exp2f(c1[c] - mn);
            const float ls = rowsum16(a + bb);
            p0[c] = a; p1[c] = bb;
            l_r[c] = l_r[c] * sc + ls;
            m_r[c] = mn;
            scale[c] = sc;
        }
        // rescale accumulators (row of component c == row of stats c)
        #pragma unroll
        for (int j = 0; j < 8; ++j)
            #pragma unroll
            for (int c = 0; c < 8; ++c) {
                accK[j][c] *= scale[c];
                accV[j][c] *= scale[c];
            }

        // ---- P (C-layout) -> A-layout via per-wave LDS staging
        unsigned short* pbuf = &sP[wave * QTILE * P_STRIDE];
        #pragma unroll
        for (int c = 0; c < 8; ++c) {
            const int row = c + 8 * half;
            const unsigned int pk = f2bf2(p0[c], p1[c]);
            pbuf[row * P_STRIDE + ln]      = (unsigned short)pk;
            pbuf[row * P_STRIDE + 16 + ln] = (unsigned short)(pk >> 16);
        }
        asm volatile("s_wait_dscnt 0" ::: "memory");   // same-wave DS RAW fence
        BfPack pa;
        {
            const int klo = half * 8;
            pa.f4[0] = *reinterpret_cast<const f32x4*>(&pbuf[ln * P_STRIDE + klo]);
            pa.f4[1] = *reinterpret_cast<const f32x4*>(&pbuf[ln * P_STRIDE + klo + 16]);
        }

        // ---- acc += P (16x32) * K_chunk / V_chunk (32x128); j-pairs so each
        //      load group feeds 4 independent WMMAs
        #pragma unroll
        for (int jb = 0; jb < 8; jb += 2) {
            const int klo = half * 8;
            const int dr0 = (16 * jb + ln) * KT_STRIDE;
            const int dr1 = (16 * (jb + 1) + ln) * KT_STRIDE;
            BfPack bk0, bv0, bk1, bv1;
            bk0.f4[0] = *reinterpret_cast<const f32x4*>(&sKT[dr0 + klo]);
            bk0.f4[1] = *reinterpret_cast<const f32x4*>(&sKT[dr0 + klo + 16]);
            bv0.f4[0] = *reinterpret_cast<const f32x4*>(&sVT[dr0 + klo]);
            bv0.f4[1] = *reinterpret_cast<const f32x4*>(&sVT[dr0 + klo + 16]);
            bk1.f4[0] = *reinterpret_cast<const f32x4*>(&sKT[dr1 + klo]);
            bk1.f4[1] = *reinterpret_cast<const f32x4*>(&sKT[dr1 + klo + 16]);
            bv1.f4[0] = *reinterpret_cast<const f32x4*>(&sVT[dr1 + klo]);
            bv1.f4[1] = *reinterpret_cast<const f32x4*>(&sVT[dr1 + klo + 16]);
            accK[jb]     = __builtin_amdgcn_wmma_f32_16x16x32_bf16(false, pa.v, false, bk0.v,
                                                                   (short)0, accK[jb], false, false);
            accV[jb]     = __builtin_amdgcn_wmma_f32_16x16x32_bf16(false, pa.v, false, bv0.v,
                                                                   (short)0, accV[jb], false, false);
            accK[jb + 1] = __builtin_amdgcn_wmma_f32_16x16x32_bf16(false, pa.v, false, bk1.v,
                                                                   (short)0, accK[jb + 1], false, false);
            accV[jb + 1] = __builtin_amdgcn_wmma_f32_16x16x32_bf16(false, pa.v, false, bv1.v,
                                                                   (short)0, accV[jb + 1], false, false);
        }
    }
    // =====================================================================

    // ---- epilogue: divide by softmax denominator, store fp32 outputs
    #pragma unroll
    for (int c = 0; c < 8; ++c) {
        const float  inv = 1.0f / l_r[c];
        const size_t row = (size_t)(qbase + c + 8 * half) * EMBED_D;
        #pragma unroll
        for (int j = 0; j < 8; ++j) {
            const int d = 16 * j + ln;
            outK[row + d] = accK[j][c] * inv;
            outV[row + d] = accV[j][c] * inv;
        }
    }
}

extern "C" void kernel_launch(void* const* d_in, const int* in_sizes, int n_in,
                              void* d_out, int out_size, void* d_ws, size_t ws_size,
                              hipStream_t stream) {
    (void)in_sizes; (void)n_in; (void)d_ws; (void)ws_size; (void)out_size;
    const float* Q = (const float*)d_in[0];   // (2,2048,128) fp32
    const float* K = (const float*)d_in[1];   // (65536,128) fp32
    const float* V = (const float*)d_in[2];   // (65536,128) fp32
    float* outK = (float*)d_out;                                 // key_vectors
    float* outV = (float*)d_out + (size_t)QROWS_TOTAL * EMBED_D; // value_vectors

    const int grid = QROWS_TOTAL / (QTILE * WAVES_PER_WG);       // 64 blocks
    lamini_attn_kernel<<<grid, WG_THREADS, 0, stream>>>(Q, K, V, outK, outV);
}
